// SlidingWindowAttention_75144747811538
// MI455X (gfx1250) — matrix operand order repaired and checked by reading, not compiled, not run on hardware
//
#include <hip/hip_runtime.h>

// ---------------- types ----------------
typedef __attribute__((ext_vector_type(16))) _Float16 v16h;
typedef __attribute__((ext_vector_type(8)))  _Float16 v8h;
typedef __attribute__((ext_vector_type(4)))  _Float16 v4h;
typedef __attribute__((ext_vector_type(8)))  float    v8f;
typedef __attribute__((ext_vector_type(4)))  float    v4f;
typedef __attribute__((ext_vector_type(4)))  int      v4i;

#define SEQ    2048
#define EMB    1024
#define NH     16
#define HD     64
#define WIN    512
#define NBATCH 2

// ---- CDNA5 async global->LDS path (guarded: falls back to VGPR copies) ----
#if __has_builtin(__builtin_amdgcn_global_load_async_to_lds_b128)
#define HAVE_ASYNC 1
#else
#define HAVE_ASYNC 0
#endif

#define AS_GLOBAL __attribute__((address_space(1)))
#define AS_LDS    __attribute__((address_space(3)))

__device__ __forceinline__ void async_cp16(_Float16* lds, const _Float16* g) {
#if HAVE_ASYNC
  // one instruction per wave: 32 lanes x 16B, LDS write bypasses VGPRs,
  // tracked by ASYNCcnt. Builtin is typed: b128 takes v4i pointers.
  _Float16* gnc = const_cast<_Float16*>(g);
  __builtin_amdgcn_global_load_async_to_lds_b128(
      (AS_GLOBAL v4i*)gnc, (AS_LDS v4i*)lds, 0, 0);
#else
  *(v8h*)lds = *(const v8h*)g;
#endif
}

__device__ __forceinline__ void wait_async() {
#if HAVE_ASYNC
#if __has_builtin(__builtin_amdgcn_s_wait_asynccnt)
  __builtin_amdgcn_s_wait_asynccnt(0);
#else
  asm volatile("s_wait_asynccnt 0x0" ::: "memory");
#endif
#endif
}

__device__ __forceinline__ v16h cat8(v8h lo, v8h hi) {
  v16h r;
#pragma unroll
  for (int i = 0; i < 8; ++i) { r[i] = lo[i]; r[i + 8] = hi[i]; }
  return r;
}

__device__ __forceinline__ v8f wmma16(v16h a, v16h b, v8f c) {
  // D = A(16x32 f16) * B(32x16 f16) + C(16x16 f32)
  return __builtin_amdgcn_wmma_f32_16x16x32_f16(false, a, false, b, (short)0, c,
                                                false, false);
}

// ---------------- f32 -> f16 convert ----------------
__global__ void cvt_f16(const float* __restrict__ in, _Float16* __restrict__ out,
                        int n4) {
  int i = blockIdx.x * blockDim.x + threadIdx.x;
  if (i >= n4) return;
  v4f x = ((const v4f*)in)[i];
  v4h y;
#pragma unroll
  for (int e = 0; e < 4; ++e) y[e] = (_Float16)x[e];
  ((v4h*)out)[i] = y;
}

// ---------------- WMMA GEMM: out[m,n] = sum_k A[m,k] * W[n,k] ----------------
// A: [M,K] f16 row-major.  W: [N,K] f16 row-major (nn.Linear layout -> W rows
// are already the B-operand columns, no transpose anywhere).
// Block: 256 threads = 8 waves; block tile 128(M) x 256(N); wave tile 64x64.
// Software pipeline: async-fill stage k+1 while WMMAs consume stage k.
template <int OUTF32>
__global__ __launch_bounds__(256) void gemm_wmma(
    const _Float16* __restrict__ A, const _Float16* __restrict__ W,
    _Float16* __restrict__ outh, float* __restrict__ outf, int M, int N, int K) {
  __shared__ _Float16 As[2][128][40];  // 32-wide K stage, pitch 40 (16B align)
  __shared__ _Float16 Bs[2][256][40];

  const int tid  = threadIdx.x;
  const int l    = tid & 31;
  const int wv   = tid >> 5;
  const int lm   = l & 15;
  const int half = l >> 4;
  const int wm   = (wv & 1) * 64;
  const int wn   = (wv >> 1) * 64;
  const int m0   = blockIdx.y * 128;
  const int n0   = blockIdx.x * 256;

  v8f acc[4][4] = {};

  auto stage = [&](int buf, int kb) {
#pragma unroll
    for (int q = 0; q < 2; ++q) {  // A tile: 128 rows x 32 f16 = 512 x 16B
      int c = tid + q * 256, r = c >> 2, sg = (c & 3) * 8;
      async_cp16(&As[buf][r][sg], &A[(size_t)(m0 + r) * K + kb + sg]);
    }
#pragma unroll
    for (int q = 0; q < 4; ++q) {  // B tile: 256 rows x 32 f16 = 1024 x 16B
      int c = tid + q * 256, r = c >> 2, sg = (c & 3) * 8;
      async_cp16(&Bs[buf][r][sg], &W[(size_t)(n0 + r) * K + kb + sg]);
    }
  };

  auto compute = [&](int buf) {
    v16h af[4], bf[4];
#pragma unroll
    for (int i = 0; i < 4; ++i) {
      // A layout: lane=M (both halves); half selects K {0-7,16-23}/{8-15,24-31}
      const _Float16* p = &As[buf][wm + i * 16 + lm][half * 8];
      af[i] = cat8(*(const v8h*)p, *(const v8h*)(p + 16));
    }
#pragma unroll
    for (int j = 0; j < 4; ++j) {
      // B layout: lane=N; half selects K {0-7,16-23}/{8-15,24-31}
      const _Float16* p = &Bs[buf][wn + j * 16 + lm][half * 8];
      bf[j] = cat8(*(const v8h*)p, *(const v8h*)(p + 16));
    }
#pragma unroll
    for (int i = 0; i < 4; ++i)
#pragma unroll
      for (int j = 0; j < 4; ++j) acc[i][j] = wmma16(af[i], bf[j], acc[i][j]);
  };

  stage(0, 0);
  wait_async();
  __syncthreads();
  for (int kb = 0; kb < K; kb += 64) {  // two pipelined 32-wide stages per iter
    if (kb + 32 < K) stage(1, kb + 32);
    compute(0);
    wait_async();
    __syncthreads();
    if (kb + 64 < K) stage(0, kb + 64);
    if (kb + 32 < K) {
      compute(1);
    }
    wait_async();
    __syncthreads();
  }

  // Epilogue. D layout: lane=N, VGPR v -> M = v + 8*half.
#pragma unroll
  for (int i = 0; i < 4; ++i)
#pragma unroll
    for (int j = 0; j < 4; ++j) {
      const int col = n0 + wn + j * 16 + lm;
#pragma unroll
      for (int v = 0; v < 8; ++v) {
        const int row = m0 + wm + i * 16 + v + 8 * half;
        if (OUTF32)
          outf[(size_t)row * N + col] = acc[i][j][v];
        else
          outh[(size_t)row * N + col] = (_Float16)acc[i][j][v];
      }
    }
}

// ---------------- Sliding-window flash attention ----------------
// Grid: (SEQ/128, NH, NBATCH). 8 waves; wave w owns queries q0..q0+15.
// Computes S' = K_tile * Q^T (keys on M) so the softmax key-reduction is
// within-lane + one shfl_xor(16); exp'd P repacks per-lane into the
// A-fragment of P*V. K/V tiles double-buffered; K fill is async.
__global__ __launch_bounds__(256) void attn_swa(
    const _Float16* __restrict__ Q, const _Float16* __restrict__ Kg,
    const _Float16* __restrict__ V, _Float16* __restrict__ O) {
  __shared__ _Float16 Ks[2][64][80];  // K tile, row-major [key][d]
  __shared__ _Float16 Vt[2][64][80];  // V tile transposed [d][key]
  __shared__ float    red[8][16];     // per-wave broadcast scratch

  const int tid  = threadIdx.x;
  const int l    = tid & 31;
  const int wv   = tid >> 5;
  const int lm   = l & 15;
  const int half = l >> 4;
  const int b    = blockIdx.z;
  const int h    = blockIdx.y;
  const int qb   = blockIdx.x * 128;
  const int q0   = qb + wv * 16;

  const size_t hoff  = (size_t)h * HD;
  const size_t bbase = (size_t)b * SEQ * EMB;

  // Q as B-fragments (lane = query), kept in registers for the whole pass.
  v16h qf[2];
#pragma unroll
  for (int ks = 0; ks < 2; ++ks) {
    const _Float16* p =
        Q + bbase + (size_t)(q0 + lm) * EMB + hoff + ks * 32 + half * 8;
    qf[ks] = cat8(*(const v8h*)p, *(const v8h*)(p + 16));
  }

  float mrun = -1e30f, lsum = 0.f;
  v8f o[4] = {};

  int jlo = qb - (WIN - 1);
  if (jlo < 0) jlo = 0;
  const int t0 = jlo >> 6;
  const int t1 = (qb + 127) >> 6;

  auto load_tile = [&](int buf, int jb) {
    const int r  = tid >> 2;
    const int sg = (tid & 3) * 16;
    const _Float16* kp = Kg + bbase + (size_t)(jb + r) * EMB + hoff + sg;
    async_cp16(&Ks[buf][r][sg], kp);
    async_cp16(&Ks[buf][r][sg + 8], kp + 8);
    const _Float16* vp = V + bbase + (size_t)(jb + r) * EMB + hoff + sg;
    v8h va = *(const v8h*)vp, vb = *(const v8h*)(vp + 8);
#pragma unroll
    for (int e = 0; e < 8; ++e) {  // transpose V into LDS
      Vt[buf][sg + e][r]     = va[e];
      Vt[buf][sg + 8 + e][r] = vb[e];
    }
  };

  load_tile(0, t0 << 6);
  wait_async();
  __syncthreads();

  for (int jt = t0; jt <= t1; ++jt) {
    const int jb  = jt << 6;
    const int buf = (jt - t0) & 1;
    if (jt < t1) load_tile(buf ^ 1, (jt + 1) << 6);  // overlap with compute

    if (jb <= q0 + 15 && jb + 63 >= q0 - (WIN - 1)) {  // wave's band only
      // S' = K * Q^T : 4 key-subtiles x 2 k-steps
      v8f s[4] = {};
#pragma unroll
      for (int mt = 0; mt < 4; ++mt)
#pragma unroll
        for (int ks = 0; ks < 2; ++ks) {
          const _Float16* p = &Ks[buf][mt * 16 + lm][ks * 32 + half * 8];
          s[mt] = wmma16(cat8(*(const v8h*)p, *(const v8h*)(p + 16)), qf[ks],
                         s[mt]);
        }

      // band mask + online softmax in exp2 domain
      const float cl2 = 0.125f * 1.4426950408889634f;  // 1/sqrt(64)*log2(e)
      const int   iq  = q0 + lm;
      float p2[4][8];
      float tmax = -1e30f;
#pragma unroll
      for (int mt = 0; mt < 4; ++mt)
#pragma unroll
        for (int v = 0; v < 8; ++v) {
          const int  j  = jb + mt * 16 + v + 8 * half;  // key index
          const bool ok = (j <= iq) && (j > iq - WIN);
          const float sv = ok ? s[mt][v] * cl2 : -__builtin_inff();
          p2[mt][v] = sv;
          tmax = fmaxf(tmax, sv);
        }
      tmax = fmaxf(tmax, __shfl_xor(tmax, 16, 32));
      const float mnew  = fmaxf(mrun, tmax);
      const float alpha = exp2f(mrun - mnew);
      mrun = mnew;

      float ts = 0.f;
#pragma unroll
      for (int mt = 0; mt < 4; ++mt)
#pragma unroll
        for (int v = 0; v < 8; ++v) {
          const float e = exp2f(p2[mt][v] - mnew);  // masked -> exp2(-inf)=0
          p2[mt][v] = e;
          ts += e;
        }
      ts += __shfl_xor(ts, 16, 32);
      lsum = lsum * alpha + ts;

      // broadcast alpha[q] to the D-layout index q = v + 8*half
      if (l < 16) red[wv][l] = alpha;
      float a8[8];
#pragma unroll
      for (int v = 0; v < 8; ++v) a8[v] = red[wv][v + 8 * half];
#pragma unroll
      for (int nt = 0; nt < 4; ++nt)
#pragma unroll
        for (int v = 0; v < 8; ++v) o[nt][v] *= a8[v];

      // P (S'-accum layout) -> A-fragment layout: pure per-lane repack
      v16h pa0, pa1;
#pragma unroll
      for (int e = 0; e < 16; ++e) {
        pa0[e] = (_Float16)(e < 8 ? p2[0][e] : p2[1][e - 8]);
        pa1[e] = (_Float16)(e < 8 ? p2[2][e] : p2[3][e - 8]);
      }
      // O += P * V : 4 d-subtiles x 2 key-chunks
#pragma unroll
      for (int nt = 0; nt < 4; ++nt)
#pragma unroll
        for (int kc = 0; kc < 2; ++kc) {
          const _Float16* p = &Vt[buf][nt * 16 + lm][kc * 32 + half * 8];
          o[nt] = wmma16(kc ? pa1 : pa0,
                         cat8(*(const v8h*)p, *(const v8h*)(p + 16)), o[nt]);
        }
    }
    wait_async();
    __syncthreads();
  }

  // final 1/l normalization + store (every query has >=1 valid key: j==i)
  if (l < 16) red[wv][l] = lsum;
  float inv8[8];
#pragma unroll
  for (int v = 0; v < 8; ++v) inv8[v] = 1.0f / red[wv][v + 8 * half];
#pragma unroll
  for (int nt = 0; nt < 4; ++nt)
#pragma unroll
    for (int v = 0; v < 8; ++v) {
      const int q = q0 + v + 8 * half;
      O[bbase + (size_t)q * EMB + hoff + nt * 16 + lm] =
          (_Float16)(o[nt][v] * inv8[v]);
    }
}

// ---------------- host launcher ----------------
extern "C" void kernel_launch(void* const* d_in, const int* in_sizes, int n_in,
                              void* d_out, int out_size, void* d_ws,
                              size_t ws_size, hipStream_t stream) {
  (void)in_sizes; (void)n_in; (void)out_size; (void)ws_size;
  const float* x  = (const float*)d_in[0];
  const float* Wq = (const float*)d_in[1];
  const float* Wk = (const float*)d_in[2];
  const float* Wv = (const float*)d_in[3];
  const float* Wo = (const float*)d_in[4];
  float* out = (float*)d_out;

  const int    M  = NBATCH * SEQ;       // 4096
  const size_t XN = (size_t)M * EMB;    // x / q / k / v / attn elements
  const size_t WN = (size_t)EMB * EMB;  // weight elements

  char*  ws  = (char*)d_ws;
  size_t off = 0;
  auto nexth = [&](size_t elems) {
    _Float16* p = (_Float16*)(ws + off);
    off += elems * sizeof(_Float16);
    return p;
  };
  _Float16* xh  = nexth(XN);
  _Float16* wqh = nexth(WN);
  _Float16* wkh = nexth(WN);
  _Float16* wvh = nexth(WN);
  _Float16* woh = nexth(WN);
  _Float16* qh  = nexth(XN);
  _Float16* kh  = nexth(XN);
  _Float16* vh  = nexth(XN);
  _Float16* ah  = nexth(XN);  // total ~48 MB of d_ws

  cvt_f16<<<(int)(XN / 4 / 256), 256, 0, stream>>>(x, xh, (int)(XN / 4));
  cvt_f16<<<(int)(WN / 4 / 256), 256, 0, stream>>>(Wq, wqh, (int)(WN / 4));
  cvt_f16<<<(int)(WN / 4 / 256), 256, 0, stream>>>(Wk, wkh, (int)(WN / 4));
  cvt_f16<<<(int)(WN / 4 / 256), 256, 0, stream>>>(Wv, wvh, (int)(WN / 4));
  cvt_f16<<<(int)(WN / 4 / 256), 256, 0, stream>>>(Wo, woh, (int)(WN / 4));

  dim3 gg(EMB / 256, M / 128);  // (4, 32)
  gemm_wmma<0><<<gg, 256, 0, stream>>>(xh, wqh, qh, nullptr, M, EMB, EMB);
  gemm_wmma<0><<<gg, 256, 0, stream>>>(xh, wkh, kh, nullptr, M, EMB, EMB);
  gemm_wmma<0><<<gg, 256, 0, stream>>>(xh, wvh, vh, nullptr, M, EMB, EMB);

  attn_swa<<<dim3(SEQ / 128, NH, NBATCH), 256, 0, stream>>>(qh, kh, vh, ah);

  gemm_wmma<1><<<gg, 256, 0, stream>>>(ah, woh, nullptr, out, M, EMB, EMB);
}
